// MambaBlock_80582176407601
// MI455X (gfx1250) — compile-verified, matrix operand order checked by reading, not run on hardware
//
#include <hip/hip_runtime.h>
#include <hip/hip_bf16.h>

#ifndef LN_EPS
#define LN_EPS 1e-5f
#endif

// ---- feature probes (evaluated separately in host/device preprocessor passes) ----
#if defined(__has_builtin)
#  if __has_builtin(__builtin_amdgcn_tensor_load_to_lds)
#    define HAVE_TDM 1
#  endif
#  if __has_builtin(__builtin_amdgcn_s_wait_tensorcnt)
#    define HAVE_TWAIT 1
#  endif
#endif

typedef __attribute__((ext_vector_type(16))) _Float16 v16h;
typedef __attribute__((ext_vector_type(8)))  _Float16 v8h;
typedef __attribute__((ext_vector_type(8)))  float    v8f;
typedef __attribute__((ext_vector_type(4)))  unsigned int u32x4;
typedef __attribute__((ext_vector_type(8)))  int          i32x8;
typedef __attribute__((ext_vector_type(4)))  int          i32x4;

#define BM 128
#define BN 64
#define BK 32
#define DSTATE 16
#define SCAN_CHUNK 64

// -------------------- weight convert + transpose:  Wt[n][k] = (f16) W[k][n] --------------------
__global__ void wt_convert_kernel(const float* __restrict__ W, _Float16* __restrict__ Wt,
                                  int K, int Nreal, int Npad) {
    long idx = (long)blockIdx.x * blockDim.x + threadIdx.x;
    long total = (long)Npad * K;
    if (idx >= total) return;
    int n = (int)(idx / K);
    int k = (int)(idx % K);
    float v = (n < Nreal) ? W[(long)k * Nreal + n] : 0.0f;
    Wt[idx] = (_Float16)v;
}

// -------------------- LayerNorm over d_model, emit f16 activations --------------------
__global__ __launch_bounds__(256)
void ln_kernel(const float* __restrict__ x, const float* __restrict__ g,
               const float* __restrict__ b, _Float16* __restrict__ xn, int dm) {
    int row = blockIdx.x;
    const float* xr = x + (long)row * dm;
    float s = 0.f, s2 = 0.f;
    for (int i = threadIdx.x; i < dm; i += blockDim.x) { float v = xr[i]; s += v; s2 += v * v; }
    __shared__ float red0[8], red1[8];
    for (int off = 16; off > 0; off >>= 1) { s += __shfl_down(s, off, 32); s2 += __shfl_down(s2, off, 32); }
    int wave = threadIdx.x >> 5, lane = threadIdx.x & 31;
    if (lane == 0) { red0[wave] = s; red1[wave] = s2; }
    __syncthreads();
    if (threadIdx.x == 0) {
        float a = 0.f, c = 0.f;
        int nw = blockDim.x >> 5;
        for (int i = 0; i < nw; ++i) { a += red0[i]; c += red1[i]; }
        red0[0] = a; red1[0] = c;
    }
    __syncthreads();
    float mu = red0[0] / dm;
    float var = red1[0] / dm - mu * mu;
    float rs = rsqrtf(var + LN_EPS);
    _Float16* o = xn + (long)row * dm;
    for (int i = threadIdx.x; i < dm; i += blockDim.x)
        o[i] = (_Float16)((xr[i] - mu) * rs * g[i] + b[i]);
}

// -------------------- WMMA fragment gather from a [rows][BK] f16 LDS tile --------------------
// 16-bit operand layout (ISA 7.12.2): lane group = lane/16; halves 0..7 -> K = grp*8..grp*8+7,
// halves 8..15 -> K = 16+grp*8 .. 16+grp*8+7. Both runs are contiguous -> two 16B LDS loads.
__device__ __forceinline__ v16h frag_from_lds(const _Float16* base, int row, int grp) {
    const _Float16* p = base + row * BK + grp * 8;
    v8h lo = *(const v8h*)p;
    v8h hi = *(const v8h*)(p + 16);
    return __builtin_shufflevector(lo, hi, 0,1,2,3,4,5,6,7,8,9,10,11,12,13,14,15);
}

#if defined(HAVE_TDM)
// -------------------- TDM: async 2D f16 tile (tile_rows x BK) from src[nrows][K] into LDS --------------------
__device__ __forceinline__ void tdm_load_tile_f16(unsigned lds_off, const _Float16* gsrc,
                                                  int K, int nrows, int tile_rows) {
    unsigned long long gaddr = (unsigned long long)(const void*)gsrc;
    u32x4 g0;
    g0[0] = 1u;                                           // count=1, user mode
    g0[1] = lds_off;                                      // lds_addr (bytes)
    g0[2] = (unsigned)(gaddr & 0xffffffffu);              // global_addr[31:0]
    g0[3] = (unsigned)((gaddr >> 32) & 0x1ffffffu)        // global_addr[56:32]
            | (2u << 30);                                 // type=2 ("image")
    i32x8 g1;
    g1[0] = 0x00010000;                                   // data_size=1 (2B), wg_mask=0
    g1[1] = (int)((K & 0xffff) << 16);                    // tensor_dim0 lo16
    g1[2] = (int)(((unsigned)K >> 16) & 0xffffu) | ((nrows & 0xffff) << 16);
    g1[3] = (int)((((unsigned)nrows >> 16) & 0xffffu) | ((unsigned)BK << 16)); // tile_dim0=BK
    g1[4] = tile_rows;                                    // tile_dim1, tile_dim2=0
    g1[5] = K;                                            // tensor_dim0_stride lo32
    g1[6] = 0;
    g1[7] = 0;
    i32x4 z4; z4[0] = 0; z4[1] = 0; z4[2] = 0; z4[3] = 0;
    i32x8 z8; z8[0] = 0; z8[1] = 0; z8[2] = 0; z8[3] = 0;
    z8[4] = 0; z8[5] = 0; z8[6] = 0; z8[7] = 0;
    __builtin_amdgcn_tensor_load_to_lds(g0, g1, z4, z4, z8, 0);
}
#endif

// -------------------- generic f16 WMMA GEMM:  C[M][Npad] = A[M][K] x Wt[Npad][K]^T --------------------
// TDM double-buffered pipeline: wave0 DMAs (A,B) tiles for iteration i+1 while all waves compute i.
// epilogue: optional bias[n], optional softplus, optional residual[row*Nreal + n]
__global__ __launch_bounds__(128)
void gemm_f16_wmma(const _Float16* __restrict__ A, const _Float16* __restrict__ Wt,
                   float* __restrict__ C, const float* __restrict__ bias,
                   const float* __restrict__ resid,
                   int M, int K, int Npad, int Nreal, int act) {
    __shared__ __align__(16) _Float16 ldsA[2 * BM * BK];   // 16 KB
    __shared__ __align__(16) _Float16 ldsB[2 * BN * BK];   // 8 KB

    const int tid  = threadIdx.x;
    const int wave = tid >> 5;
    const int lane = tid & 31;
    const int grp  = lane >> 4;       // 0 or 1
    const int lr   = lane & 15;       // 0..15
    const int bm   = blockIdx.x * BM;
    const int bn   = blockIdx.y * BN;

    v8f acc[2][4] = {};
    const int nIter = K / BK;

#if defined(HAVE_TDM)
    const unsigned ldsA_off = (unsigned)(unsigned long long)(void*)&ldsA[0];
    const unsigned ldsB_off = (unsigned)(unsigned long long)(void*)&ldsB[0];
    if (wave == 0) {   // prologue: fill buffer 0
        tdm_load_tile_f16(ldsA_off, A + (long)bm * K, K, M, BM);
        tdm_load_tile_f16(ldsB_off, Wt + (long)bn * K, K, Npad, BN);
    }
#endif

    for (int i = 0; i < nIter; ++i) {
        const int cur = i & 1;
#if defined(HAVE_TDM)
        if (wave == 0) {
#  if defined(HAVE_TWAIT)
            __builtin_amdgcn_s_wait_tensorcnt(0);
#  else
            asm volatile("s_wait_tensorcnt 0x0" ::: "memory");
#  endif
        }
        __syncthreads();                       // buf[cur] ready; prior compute done by all waves
        if (wave == 0 && (i + 1) < nIter) {    // DMA next tiles into buf[cur^1]
            const int nxt = cur ^ 1;
            const int k1 = (i + 1) * BK;
            tdm_load_tile_f16(ldsA_off + (unsigned)(nxt * BM * BK * 2),
                              A + (long)bm * K + k1, K, M, BM);
            tdm_load_tile_f16(ldsB_off + (unsigned)(nxt * BN * BK * 2),
                              Wt + (long)bn * K + k1, K, Npad, BN);
        }
#else
        // fallback: VGPR staging of both tiles (single-buffered)
        __syncthreads();
        {
            const int k0 = i * BK;
            for (int r = tid; r < BM + BN; r += 128) {
                const uint4* src;
                uint4* dst;
                if (r < BM) {
                    src = (const uint4*)(A + (long)(bm + r) * K + k0);
                    dst = (uint4*)(ldsA + cur * BM * BK + r * BK);
                } else {
                    src = (const uint4*)(Wt + (long)(bn + (r - BM)) * K + k0);
                    dst = (uint4*)(ldsB + cur * BN * BK + (r - BM) * BK);
                }
                dst[0] = src[0]; dst[1] = src[1]; dst[2] = src[2]; dst[3] = src[3];
            }
        }
        __syncthreads();
#endif
        // ---- compute: each wave owns 32 rows x 64 cols of C ----
        const _Float16* lA = ldsA + cur * BM * BK;
        const _Float16* lB = ldsB + cur * BN * BK;
        v16h af0 = frag_from_lds(lA, wave * 32 + lr, grp);
        v16h af1 = frag_from_lds(lA, wave * 32 + 16 + lr, grp);
#pragma unroll
        for (int c = 0; c < 4; ++c) {
            v16h bf = frag_from_lds(lB, c * 16 + lr, grp);
            acc[0][c] = __builtin_amdgcn_wmma_f32_16x16x32_f16(
                false, af0, false, bf, (short)0, acc[0][c], false, false);
            acc[1][c] = __builtin_amdgcn_wmma_f32_16x16x32_f16(
                false, af1, false, bf, (short)0, acc[1][c], false, false);
        }
        // next iteration's top barrier provides the read-complete fence
    }

    // ---- epilogue: C/D layout = VGPR r: lanes0-15 -> M=r, lanes16-31 -> M=r+8; N=lane%16 ----
#pragma unroll
    for (int s = 0; s < 2; ++s) {
        const int row_lo = bm + wave * 32 + s * 16 + grp * 8;
#pragma unroll
        for (int c = 0; c < 4; ++c) {
            int col = bn + c * 16 + lr;
            if (col < Nreal) {
                float bv = bias ? bias[col] : 0.0f;
#pragma unroll
                for (int r = 0; r < 8; ++r) {
                    int row = row_lo + r;
                    float v = acc[s][c][r] + bv;
                    if (act == 1) v = (v > 20.0f) ? v : log1pf(__expf(v));   // softplus
                    if (resid) v += resid[(long)row * Nreal + col];
                    C[(long)row * Npad + col] = v;
                }
            }
        }
    }
}

// -------------------- causal depthwise conv1d (d_conv=4) + bias + SiLU --------------------
__global__ __launch_bounds__(256)
void conv_silu_kernel(const float* __restrict__ xz, const float* __restrict__ cw,
                      const float* __restrict__ cb, float* __restrict__ xact_f,
                      _Float16* __restrict__ xact_h, int L, int di, int twoDi) {
    long idx = (long)blockIdx.x * blockDim.x + threadIdx.x;   // over B*L*di
    int d = (int)(idx % di);
    long row = idx / di;                                      // b*L + l
    int l = (int)(row % L);
    float acc = cb[d];
#pragma unroll
    for (int j = 0; j < 4; ++j) {
        int ll = l - 3 + j;
        if (ll >= 0) acc += cw[d * 4 + j] * xz[(row - 3 + j) * (long)twoDi + d];
    }
    float s = acc / (1.0f + __expf(-acc));                    // silu
    xact_f[idx] = s;
    xact_h[idx] = (_Float16)s;
}

// -------------------- selective scan: one (b,d) channel per lane, h[16] in VGPRs --------------------
__global__ __launch_bounds__(256)
void scan_kernel(const float* __restrict__ dt, const float* __restrict__ xdbl /*[M][64]*/,
                 const float* __restrict__ xact, const float* __restrict__ A_log,
                 const float* __restrict__ Dp, float* __restrict__ y, int L, int di) {
    int ch = blockIdx.x * blockDim.x + threadIdx.x;           // 0..B*di-1
    int b = ch / di;
    int d = ch % di;
    float Arow[DSTATE];
#pragma unroll
    for (int n = 0; n < DSTATE; ++n) Arow[n] = -__expf(A_log[d * DSTATE + n]);
    float h[DSTATE] = {};
    float Dv = Dp[d];
    __shared__ float sBC[SCAN_CHUNK][32];
    long base = (long)b * L;
    for (int l0 = 0; l0 < L; l0 += SCAN_CHUNK) {
        __syncthreads();
        for (int i = threadIdx.x; i < SCAN_CHUNK * 32; i += blockDim.x) {
            int r = i >> 5, c = i & 31;
            sBC[r][c] = xdbl[(base + l0 + r) * 64 + c];
        }
        __syncthreads();
        if (l0 + SCAN_CHUNK < L)   // hint next chunk of dt into cache -> global_prefetch_b8
            __builtin_prefetch(dt + (base + l0 + SCAN_CHUNK) * di + d, 0, 1);
        for (int r = 0; r < SCAN_CHUNK; ++r) {
            long row = base + l0 + r;
            float dtv = dt[row * di + d];
            float xv  = xact[row * di + d];
            float out = 0.0f;
#pragma unroll
            for (int n = 0; n < DSTATE; ++n) {
                float Ab = __expf(dtv * Arow[n]);             // v_exp_f32
                h[n] = Ab * h[n] + (dtv * sBC[r][n]) * xv;
                out += h[n] * sBC[r][16 + n];
            }
            y[row * di + d] = out + Dv * xv;
        }
    }
}

// -------------------- gating: y_gated = f16( y_scan * silu(z) ) --------------------
__global__ __launch_bounds__(256)
void gate_kernel(const float* __restrict__ yscan, const float* __restrict__ xz,
                 _Float16* __restrict__ ygate, int di, int twoDi) {
    long idx = (long)blockIdx.x * blockDim.x + threadIdx.x;   // over B*L*di
    int d = (int)(idx % di);
    long row = idx / di;
    float z = xz[row * (long)twoDi + di + d];
    float sz = z / (1.0f + __expf(-z));
    ygate[idx] = (_Float16)(yscan[idx] * sz);
}

// ==================== launch ====================
extern "C" void kernel_launch(void* const* d_in, const int* in_sizes, int n_in,
                              void* d_out, int out_size, void* d_ws, size_t ws_size,
                              hipStream_t stream) {
    (void)in_sizes; (void)n_in; (void)out_size; (void)ws_size;
    const int B = 2, L = 2048, DM = 1024, DI = 2048;
    const int M = B * L;                  // 4096
    const int TWO_DI = 2 * DI;            // 4096
    const int NXP_PAD = 64;               // w_xproj cols padded 32 -> 64

    const float* x      = (const float*)d_in[0];
    const float* w_in   = (const float*)d_in[1];
    const float* conv_w = (const float*)d_in[2];
    const float* conv_b = (const float*)d_in[3];
    const float* w_xp   = (const float*)d_in[4];
    const float* w_dt   = (const float*)d_in[5];
    const float* b_dt   = (const float*)d_in[6];
    const float* A_log  = (const float*)d_in[7];
    const float* Dp     = (const float*)d_in[8];
    const float* w_out  = (const float*)d_in[9];
    const float* ln_g   = (const float*)d_in[10];
    const float* ln_b   = (const float*)d_in[11];
    float* out = (float*)d_out;

    // workspace carve-up
    char* ws = (char*)d_ws;
    size_t off = 0;
    auto alloc = [&](size_t bytes) { void* p = ws + off; off += (bytes + 255) & ~size_t(255); return p; };
    _Float16* wt_in  = (_Float16*)alloc((size_t)TWO_DI * DM * 2);   // [4096][1024]
    _Float16* wt_dt  = (_Float16*)alloc((size_t)DI * DI * 2);       // [2048][2048]
    _Float16* wt_xp  = (_Float16*)alloc((size_t)NXP_PAD * DI * 2);  // [64][2048]
    _Float16* wt_out = (_Float16*)alloc((size_t)DM * DI * 2);       // [1024][2048]
    _Float16* xn_h   = (_Float16*)alloc((size_t)M * DM * 2);
    float*    xz     = (float*)   alloc((size_t)M * TWO_DI * 4);
    float*    xact_f = (float*)   alloc((size_t)M * DI * 4);
    _Float16* xact_h = (_Float16*)alloc((size_t)M * DI * 2);
    float*    xdbl   = (float*)   alloc((size_t)M * NXP_PAD * 4);
    float*    dtbuf  = (float*)   alloc((size_t)M * DI * 4);
    float*    yscan  = (float*)   alloc((size_t)M * DI * 4);
    _Float16* ygate  = (_Float16*)alloc((size_t)M * DI * 2);

    // 1) convert+transpose weights to f16 [N][K]
    {
        long t;
        t = (long)TWO_DI * DM;
        wt_convert_kernel<<<(int)((t + 255) / 256), 256, 0, stream>>>(w_in, wt_in, DM, TWO_DI, TWO_DI);
        t = (long)DI * DI;
        wt_convert_kernel<<<(int)((t + 255) / 256), 256, 0, stream>>>(w_dt, wt_dt, DI, DI, DI);
        t = (long)NXP_PAD * DI;
        wt_convert_kernel<<<(int)((t + 255) / 256), 256, 0, stream>>>(w_xp, wt_xp, DI, 32, NXP_PAD);
        t = (long)DM * DI;
        wt_convert_kernel<<<(int)((t + 255) / 256), 256, 0, stream>>>(w_out, wt_out, DI, DM, DM);
    }

    // 2) layernorm -> xn (f16)
    ln_kernel<<<M, 256, 0, stream>>>(x, ln_g, ln_b, xn_h, DM);

    // 3) xz = xn @ w_in   [4096x1024]x[1024x4096]
    gemm_f16_wmma<<<dim3(M / BM, TWO_DI / BN), 128, 0, stream>>>(
        xn_h, wt_in, xz, nullptr, nullptr, M, DM, TWO_DI, TWO_DI, 0);

    // 4) causal depthwise conv + SiLU -> xact (f32 + f16)
    {
        long t = (long)M * DI;
        conv_silu_kernel<<<(int)((t + 255) / 256), 256, 0, stream>>>(
            xz, conv_w, conv_b, xact_f, xact_h, L, DI, TWO_DI);
    }

    // 5) x_dbl = xact @ w_xproj  (N padded to 64)
    gemm_f16_wmma<<<dim3(M / BM, NXP_PAD / BN), 128, 0, stream>>>(
        xact_h, wt_xp, xdbl, nullptr, nullptr, M, DI, NXP_PAD, 32, 0);

    // 6) dt = softplus(xact @ w_dt + b_dt)
    gemm_f16_wmma<<<dim3(M / BM, DI / BN), 128, 0, stream>>>(
        xact_h, wt_dt, dtbuf, b_dt, nullptr, M, DI, DI, DI, 1);

    // 7) selective scan (+ D * x)
    scan_kernel<<<(B * DI) / 256, 256, 0, stream>>>(dtbuf, xdbl, xact_f, A_log, Dp, yscan, L, DI);

    // 8) gate with silu(z)
    {
        long t = (long)M * DI;
        gate_kernel<<<(int)((t + 255) / 256), 256, 0, stream>>>(yscan, xz, ygate, DI, TWO_DI);
    }

    // 9) out = ygate @ w_out + residual(x)
    gemm_f16_wmma<<<dim3(M / BM, DM / BN), 128, 0, stream>>>(
        ygate, wt_out, out, nullptr, x, M, DI, DM, DM, 0);
}
